// MultiheadFlashAttnSWinFVDB_16819091931421
// MI455X (gfx1250) — matrix operand order repaired and checked by reading, not compiled, not run on hardware
//
#include <hip/hip_runtime.h>
#include <hip/hip_bf16.h>

// ---------------- problem constants ----------------
#define EMBD   768
#define NHEAD  16
#define HDIM   48          // EMBD / NHEAD
#define GRID_N 48
#define WINSZ  4
#define NTOK   (GRID_N*GRID_N*GRID_N)   // 110592
#define GW     (GRID_N/WINSZ)           // 12
#define NWIN   (GW*GW*GW)               // 1728
#define WLEN   (WINSZ*WINSZ*WINSZ)      // 64
#define QKVC   (3*EMBD)                 // 2304
#define BS_LD  56                       // LDS B-panel row stride (halves): 112B = 28 banks,
                                        // (28*n)%64 distinct for n=0..15 -> conflict-free b128

typedef _Float16 half_t;
typedef __attribute__((ext_vector_type(16))) _Float16 v16h;
typedef __attribute__((ext_vector_type(8)))  float    v8f;

static __device__ __forceinline__ v8f wmma_f16(v16h a, v16h b, v8f c) {
  // v_wmma_f32_16x16x32_f16  (neg_a, A, neg_b, B, c_mod, C, reuse_a, reuse_b)
  return __builtin_amdgcn_wmma_f32_16x16x32_f16(false, a, false, b, (short)0, c,
                                                false, false);
}

// A-matrix 16x32 f16 fragment (row-major source, leading dim lda).
// ISA layout: lanes 0-15 = rows, VGPR p<4 -> K = 2p (+lane-half*8),
// VGPR p>=4 -> K = 16 + 2(p-4) (+lane-half*8). kvalid zero-pads K >= kvalid.
static __device__ __forceinline__ v16h frag_a(const half_t* __restrict__ A,
                                              int lda, int kvalid) {
  const int l  = threadIdx.x & 31;
  const int m  = l & 15;
  const int kh = (l >> 4) << 3;
  v16h a;
#pragma unroll
  for (int p = 0; p < 8; ++p) {
    const int kb = ((p & 4) << 2) + kh + ((p & 3) << 1);
    half_t e0 = (half_t)0.0f, e1 = (half_t)0.0f;
    if (kb < kvalid) { e0 = A[m * lda + kb]; e1 = A[m * lda + kb + 1]; }
    a[2 * p] = e0; a[2 * p + 1] = e1;
  }
  return a;
}

// B-matrix 32x16 f16 fragment, loaded from BT where BT[n][k] = B[k][n]
// (contraction contiguous per lane). Same K order as frag_a: v<4 -> K = 2v (+lane-half*8), v>=4 -> K = 16 + 2(v-4) (+lane-half*8).
static __device__ __forceinline__ v16h frag_b(const half_t* BT, int ldbt,
                                              int kvalid) {
  const int l  = threadIdx.x & 31;
  const int n  = l & 15;
  const int kh = (l >> 4) << 3;
  v16h b;
#pragma unroll
  for (int v = 0; v < 8; ++v) {
    const int kb = ((v & 4) << 2) + kh + ((v & 3) << 1);
    half_t e0 = (half_t)0.0f, e1 = (half_t)0.0f;
    if (kb < kvalid) { e0 = BT[n * ldbt + kb]; e1 = BT[n * ldbt + kb + 1]; }
    b[2 * v] = e0; b[2 * v + 1] = e1;
  }
  return b;
}

// ---------------- kernel 1: stable window bucketing (deterministic) --------
__global__ void build_order_kernel(const int* __restrict__ ijk,
                                   int* __restrict__ order,
                                   int* __restrict__ slot) {
  const int w = blockIdx.x * blockDim.x + threadIdx.x;
  if (w >= NWIN) return;
  int cnt = 0;
  for (int n = 0; n < NTOK; ++n) {
    const int i = ijk[3 * n + 0] >> 2;
    const int j = ijk[3 * n + 1] >> 2;
    const int k = ijk[3 * n + 2] >> 2;
    const int wid = (i * GW + j) * GW + k;
    if (wid == w) {
      order[w * WLEN + cnt] = n;   // windowed row -> token (== inv perm)
      slot[n] = w * WLEN + cnt;    // token -> windowed row
      ++cnt;
    }
  }
}

// ---------------- kernel 2: f32 -> f16 transpose (weights) -----------------
__global__ void transpose_f16_kernel(const float* __restrict__ w,
                                     half_t* __restrict__ wt,
                                     int rows, int cols) {
  const int idx = blockIdx.x * blockDim.x + threadIdx.x;
  if (idx >= rows * cols) return;
  const int r = idx / cols, c = idx - r * cols;
  wt[(size_t)c * rows + r] = (half_t)w[idx];
}

// ---------------- kernel 3: f32 -> f16 copy (activations) ------------------
__global__ void tof16_kernel(const float* __restrict__ src,
                             half_t* __restrict__ dst, int total) {
  const int idx = blockIdx.x * blockDim.x + threadIdx.x;
  if (idx < total) dst[idx] = (half_t)src[idx];
}

// Stage a 64-col x 32-k B panel (BT layout: [n][k]) into LDS, one uint4/thread.
static __device__ __forceinline__ void stage_b_panel(
    half_t (*Bs)[BS_LD], const half_t* __restrict__ wT, int col0, int k0) {
  const int n  = threadIdx.x >> 2;         // 0..63
  const int ko = (threadIdx.x & 3) << 3;   // 0,8,16,24 halves
  const uint4 v = *reinterpret_cast<const uint4*>(
      wT + (size_t)(col0 + n) * EMBD + k0 + ko);
  *reinterpret_cast<uint4*>(&Bs[n][ko]) = v;
}

// ---------------- kernel 4: QKV GEMM + bias + RoPE3D + window scatter ------
// Block = 8 waves x same 64-col strip, 8 row tiles; B panel shared via LDS.
__global__ __launch_bounds__(256) void qkv_gemm_kernel(
    const half_t* __restrict__ xh, const half_t* __restrict__ wT,
    const float* __restrict__ bias, const int* __restrict__ ijk,
    const int* __restrict__ slot,
    half_t* __restrict__ Qw, half_t* __restrict__ Kw, half_t* __restrict__ Vt) {
  __shared__ __align__(16) half_t Bs[64][BS_LD];
  const int lane = threadIdx.x & 31;
  const int wave = threadIdx.x >> 5;
  const int ncs  = QKVC / 64;          // 36 column strips
  const int cs   = blockIdx.x % ncs;
  const int rb   = blockIdx.x / ncs;   // 0..863 row blocks (8 tiles each)
  const int rt   = rb * 8 + wave;      // this wave's 16-row tile
  const half_t* Arow = xh + (size_t)rt * 16 * EMBD;

  v8f z = {0.f,0.f,0.f,0.f,0.f,0.f,0.f,0.f};
  v8f acc[4] = {z, z, z, z};
  for (int k0 = 0; k0 < EMBD; k0 += 32) {
    stage_b_panel(Bs, wT, cs * 64, k0);
    __syncthreads();
    __builtin_prefetch(Arow + k0 + 128, 0, 1);   // global_prefetch_b8
    const v16h a = frag_a(Arow + k0, EMBD, 32);
#pragma unroll
    for (int t = 0; t < 4; ++t)
      acc[t] = wmma_f16(a, frag_b(&Bs[t * 16][0], BS_LD, 32), acc[t]);
    __syncthreads();
  }

  // Epilogue: C layout row = r + 8*(lane>>4), col = lane&15.
  const float l2r = 13.287712379549449f / 8.0f;  // log2(10000)/8
  const int hi16 = lane >> 4;
  const int nn   = lane & 15;
#pragma unroll
  for (int t = 0; t < 4; ++t) {
    const int c    = cs * 64 + t * 16 + nn;
    const int sec  = c / EMBD;          // 0=q 1=k 2=v (uniform per tile)
    const int cc   = c - sec * EMBD;
    const int head = cc / HDIM;
    const int ch   = cc - head * HDIM;  // 0..47
    const int axis = ch >> 4;           // rope axis chunk
    const int tt   = ch & 15;
    const int f    = tt & 7;
    const bool ishi = tt >= 8;
    const float bv   = bias[c];
    const float invf = exp2f(-(float)f * l2r);
#pragma unroll
    for (int r = 0; r < 8; ++r) {
      const int row = rt * 16 + r + 8 * hi16;   // token index
      float val = acc[t][r] + bv;
      const int s   = slot[row];
      const int wi  = s >> 6;
      const int loc = s & 63;
      if (sec < 2) {                      // rope for q and k
        const int coord = ijk[3 * row + axis];
        float sn, cn;
        __sincosf((float)coord * invf, &sn, &cn);
        const float partner = __shfl_xor(val, 8, 32);  // pair column +-8
        const float res = ishi ? (partner * sn + val * cn)
                               : (val * cn - partner * sn);
        const size_t o = (((size_t)wi * NHEAD + head) * WLEN + loc) * HDIM + ch;
        if (sec == 0) Qw[o] = (half_t)res; else Kw[o] = (half_t)res;
      } else {                            // V stored transposed (d, key)
        const size_t o = (((size_t)wi * NHEAD + head) * HDIM + ch) * WLEN + loc;
        Vt[o] = (half_t)val;
      }
    }
  }
}

// ---------------- kernel 5: 64x64 windowed attention -----------------------
// One block per (window, head); 4 waves x 16 query rows each.
__global__ __launch_bounds__(128) void attn_kernel(
    const half_t* __restrict__ Qw, const half_t* __restrict__ Kw,
    const half_t* __restrict__ Vt, half_t* __restrict__ Ow) {
  __shared__ half_t P[4][16][64];
  const int wh   = blockIdx.x;           // w*NHEAD + h
  const int wave = threadIdx.x >> 5;
  const int lane = threadIdx.x & 31;
  const half_t* Qb = Qw + (size_t)wh * WLEN * HDIM + wave * 16 * HDIM;
  const half_t* Kb = Kw + (size_t)wh * WLEN * HDIM;
  const half_t* Vb = Vt + (size_t)wh * HDIM * WLEN;

  v8f z = {0.f,0.f,0.f,0.f,0.f,0.f,0.f,0.f};
  // S = Q * K^T : D=48 as K-steps 32 + 16 (zero-padded fragments)
  v8f s[4] = {z, z, z, z};
  const v16h a0 = frag_a(Qb, HDIM, 32);
  const v16h a1 = frag_a(Qb + 32, HDIM, 16);
#pragma unroll
  for (int t = 0; t < 4; ++t) {
    s[t] = wmma_f16(a0, frag_b(Kb + (size_t)t * 16 * HDIM, HDIM, 32), s[t]);
    s[t] = wmma_f16(a1, frag_b(Kb + (size_t)t * 16 * HDIM + 32, HDIM, 16), s[t]);
  }

  const float scale = 0.14433756729740643f;  // 48^-0.5
  const int nn = lane & 15, hb = lane >> 4;
#pragma unroll
  for (int r = 0; r < 8; ++r) {
    float v0 = s[0][r]*scale, v1 = s[1][r]*scale, v2 = s[2][r]*scale, v3 = s[3][r]*scale;
    float m = fmaxf(fmaxf(v0, v1), fmaxf(v2, v3));
#pragma unroll
    for (int msk = 8; msk >= 1; msk >>= 1) m = fmaxf(m, __shfl_xor(m, msk, 32));
    float e0 = expf(v0 - m), e1 = expf(v1 - m), e2 = expf(v2 - m), e3 = expf(v3 - m);
    float sum = e0 + e1 + e2 + e3;
#pragma unroll
    for (int msk = 8; msk >= 1; msk >>= 1) sum += __shfl_xor(sum, msk, 32);
    const float inv = 1.0f / sum;
    const int row = r + 8 * hb;
    P[wave][row][ 0 + nn] = (half_t)(e0 * inv);
    P[wave][row][16 + nn] = (half_t)(e1 * inv);
    P[wave][row][32 + nn] = (half_t)(e2 * inv);
    P[wave][row][48 + nn] = (half_t)(e3 * inv);
  }
  __syncthreads();

  // O = P * V : keys=64 as 2 K-steps; V already (d, key) so B loads contiguous
  v8f o[3] = {z, z, z};
  const half_t* Pl = &P[wave][0][0];
#pragma unroll
  for (int kk = 0; kk < 64; kk += 32) {
    const v16h ap = frag_a(Pl + kk, 64, 32);
#pragma unroll
    for (int t = 0; t < 3; ++t)
      o[t] = wmma_f16(ap, frag_b(Vb + (size_t)t * 16 * 64 + kk, 64, 32), o[t]);
  }

  const int w = wh / NHEAD, h = wh % NHEAD;
#pragma unroll
  for (int t = 0; t < 3; ++t)
#pragma unroll
    for (int r = 0; r < 8; ++r) {
      const int row = wave * 16 + r + 8 * hb;
      const int d   = t * 16 + nn;
      Ow[((size_t)w * WLEN + row) * EMBD + h * HDIM + d] = (half_t)o[t][r];
    }
}

// ---------------- kernel 6: output GEMM + bias + inverse scatter -----------
__global__ __launch_bounds__(256) void out_gemm_kernel(
    const half_t* __restrict__ Ow, const half_t* __restrict__ wT,
    const float* __restrict__ bias, const int* __restrict__ order,
    float* __restrict__ out) {
  __shared__ __align__(16) half_t Bs[64][BS_LD];
  const int lane = threadIdx.x & 31;
  const int wave = threadIdx.x >> 5;
  const int ncs  = EMBD / 64;          // 12
  const int cs   = blockIdx.x % ncs;
  const int rb   = blockIdx.x / ncs;
  const int rt   = rb * 8 + wave;
  const half_t* Arow = Ow + (size_t)rt * 16 * EMBD;

  v8f z = {0.f,0.f,0.f,0.f,0.f,0.f,0.f,0.f};
  v8f acc[4] = {z, z, z, z};
  for (int k0 = 0; k0 < EMBD; k0 += 32) {
    stage_b_panel(Bs, wT, cs * 64, k0);
    __syncthreads();
    __builtin_prefetch(Arow + k0 + 128, 0, 1);
    const v16h a = frag_a(Arow + k0, EMBD, 32);
#pragma unroll
    for (int t = 0; t < 4; ++t)
      acc[t] = wmma_f16(a, frag_b(&Bs[t * 16][0], BS_LD, 32), acc[t]);
    __syncthreads();
  }
  const int nn = lane & 15, hb = lane >> 4;
#pragma unroll
  for (int t = 0; t < 4; ++t) {
    const int c = cs * 64 + t * 16 + nn;
    const float bv = bias[c];
#pragma unroll
    for (int r = 0; r < 8; ++r) {
      const int grow = rt * 16 + r + 8 * hb;      // windowed row
      const int tok  = order[grow];               // original token
      out[(size_t)tok * EMBD + c] = acc[t][r] + bv;
    }
  }
}

// ---------------- host launcher --------------------------------------------
extern "C" void kernel_launch(void* const* d_in, const int* in_sizes, int n_in,
                              void* d_out, int out_size, void* d_ws, size_t ws_size,
                              hipStream_t stream) {
  const float* x     = (const float*)d_in[0];
  const int*   ijk   = (const int*)  d_in[1];
  const float* w_qkv = (const float*)d_in[2];
  const float* b_qkv = (const float*)d_in[3];
  const float* w_out = (const float*)d_in[4];
  const float* b_out = (const float*)d_in[5];
  float* out = (float*)d_out;

  char* ws = (char*)d_ws;
  size_t off = 0;
  auto carve = [&](size_t bytes) -> void* {
    void* p = ws + off;
    off = (off + bytes + 255) & ~(size_t)255;
    return p;
  };
  const size_t qk_bytes = (size_t)NWIN * NHEAD * WLEN * HDIM * sizeof(half_t);
  half_t* xh     = (half_t*)carve((size_t)NTOK * EMBD * sizeof(half_t));
  half_t* wqkvT  = (half_t*)carve((size_t)QKVC * EMBD * sizeof(half_t));
  half_t* woutT  = (half_t*)carve((size_t)EMBD * EMBD * sizeof(half_t));
  half_t* Qw     = (half_t*)carve(qk_bytes);
  half_t* Kw     = (half_t*)carve(qk_bytes);
  half_t* Vt     = (half_t*)carve(qk_bytes);
  half_t* Ow     = (half_t*)carve((size_t)NTOK * EMBD * sizeof(half_t));
  int*    order  = (int*)   carve((size_t)NTOK * sizeof(int));
  int*    slot   = (int*)   carve((size_t)NTOK * sizeof(int));

  build_order_kernel<<<(NWIN + 63) / 64, 64, 0, stream>>>(ijk, order, slot);

  {
    int tot = EMBD * QKVC;
    transpose_f16_kernel<<<(tot + 255) / 256, 256, 0, stream>>>(w_qkv, wqkvT, EMBD, QKVC);
  }
  {
    int tot = EMBD * EMBD;
    transpose_f16_kernel<<<(tot + 255) / 256, 256, 0, stream>>>(w_out, woutT, EMBD, EMBD);
  }
  {
    int tot = NTOK * EMBD;
    tof16_kernel<<<(tot + 255) / 256, 256, 0, stream>>>(x, xh, tot);
  }

  // 864 row blocks (8 tiles of 16 rows) x 36 col strips
  qkv_gemm_kernel<<<(NTOK / 128) * (QKVC / 64), 256, 0, stream>>>(
      xh, wqkvT, b_qkv, ijk, slot, Qw, Kw, Vt);

  attn_kernel<<<NWIN * NHEAD, 128, 0, stream>>>(Qw, Kw, Vt, Ow);

  // 864 row blocks x 12 col strips
  out_gemm_kernel<<<(NTOK / 128) * (EMBD / 64), 256, 0, stream>>>(
      Ow, woutT, b_out, order, out);
}
